// EquivariantBlock_71846212927516
// MI455X (gfx1250) — matrix operand order repaired and checked by reading, not compiled, not run on hardware
//
#include <hip/hip_runtime.h>
#include <math.h>

// ---------------------------------------------------------------------------
// EGNN EquivariantBlock for MI455X (gfx1250, wave32, WMMA).
// Heavy GEMMs (edge MLPs 258->128->128 over 600K edges, node MLPs over 50K
// nodes) run on v_wmma_f32_16x16x32_bf16 with bf16 operands / f32 accum.
// h table (25.6MB) fits in 192MB L2 -> gathers/atomics are L2-resident, so
// the kernel is compute-bound and the matrix pipe is the right target.
// Round 3: fast-rcp silu/sigmoid (no precise-div chain), packed 8B bf16
// gather stores, attention/c3 weight vectors staged in LDS.
// ---------------------------------------------------------------------------

#define HID   128
#define TILE  64
#define NORMF 100.0f

typedef __bf16 bf16_t;
typedef bf16_t v16bf __attribute__((ext_vector_type(16)));
typedef float  v8f   __attribute__((ext_vector_type(8)));

struct alignas(16) U4 { unsigned int x, y, z, w; };
struct alignas(8)  U2 { unsigned int x, y; };
union alignas(32) Frag { v16bf v; unsigned short u[16]; U4 q[2]; };

__device__ __forceinline__ unsigned short f2bf(float f) {
  unsigned int u = __float_as_uint(f);
  u += 0x7FFFu + ((u >> 16) & 1u);      // round-to-nearest-even
  return (unsigned short)(u >> 16);
}
__device__ __forceinline__ unsigned int f2bf2(float lo, float hi) {
  return (unsigned int)f2bf(lo) | ((unsigned int)f2bf(hi) << 16);
}
__device__ __forceinline__ float fast_rcp(float x) {
#if __has_builtin(__builtin_amdgcn_rcpf)
  return __builtin_amdgcn_rcpf(x);      // v_rcp_f32
#else
  return 1.0f / x;
#endif
}
__device__ __forceinline__ float sigmf(float x) { return fast_rcp(1.0f + __expf(-x)); }
__device__ __forceinline__ float siluf(float x) { return x * sigmf(x); }

// A layout (ISA 7.12.2, 16-bit 16x32): lane L -> M=L%16, half=L/16;
//   elem e<8 -> K = 32*kb + 8*half + e ; e>=8 -> K = 32*kb + 8*half + 8 + e.
// B layout (32x16): lane L -> N = n0 + L%16 ; elem e -> K same as A layout.
// All fragment addresses are 16B aligned -> ds_load_b128.
template<int KBLKS, int LDA>
__device__ __forceinline__ void load_afrags(const unsigned short* A, int mBase,
                                            int lane, Frag* afr) {
  const int half = lane >> 4;
  const unsigned short* arow = A + (mBase + (lane & 15)) * LDA + 8 * half;
#pragma unroll
  for (int kb = 0; kb < KBLKS; ++kb) {
    afr[kb].q[0] = *(const U4*)(arow);
    afr[kb].q[1] = *(const U4*)(arow + 16);
    arow += 32;
  }
}

template<int KBLKS, int LDB>
__device__ __forceinline__ v8f wmma_row(const Frag* afr, const unsigned short* B,
                                        int n0, int lane, v8f acc) {
  const unsigned short* brow = B + (8 * (lane >> 4)) * LDB + n0 + (lane & 15);
#pragma unroll
  for (int kb = 0; kb < KBLKS; ++kb) {
    Frag b;
#pragma unroll
    for (int s = 0; s < 16; ++s) b.u[s] = brow[(s < 8 ? s : s + 8) * LDB];
    acc = __builtin_amdgcn_wmma_f32_16x16x32_bf16(false, afr[kb].v, false, b.v,
                                                  (short)0, acc, false, false);
    brow += 32 * LDB;
  }
  return acc;
}

// ---------------------------------------------------------------------------
// fp32 -> bf16 weight conversion
__global__ void f2bf_kernel(const float* __restrict__ s,
                            unsigned short* __restrict__ d, int n) {
  int i = blockIdx.x * blockDim.x + threadIdx.x;
  if (i < n) d[i] = f2bf(s[i]);
}

// Edge geometry: radial + normalized coord_diff
__global__ void prep_edges_kernel(const float* __restrict__ x,
                                  const int* __restrict__ eidx,
                                  float* __restrict__ radial,
                                  float* __restrict__ cdiff, int nEdges) {
  int e = blockIdx.x * blockDim.x + threadIdx.x;
  if (e >= nEdges) return;
  int r = eidx[e], c = eidx[nEdges + e];
  float dx = x[r * 3 + 0] - x[c * 3 + 0];
  float dy = x[r * 3 + 1] - x[c * 3 + 1];
  float dz = x[r * 3 + 2] - x[c * 3 + 2];
  float rad = dx * dx + dy * dy + dz * dz;
  float inv = 1.0f / sqrtf(rad + 1e-8f);
  radial[e] = rad;
  cdiff[e * 3 + 0] = dx * inv;
  cdiff[e * 3 + 1] = dy * inv;
  cdiff[e * 3 + 2] = dz * inv;
}

// ---------------------------------------------------------------------------
// GCL edge kernel: mij = silu(e2(silu(e1([h_r|h_c|ea])))), att = sigmoid(...),
// scatter agg[row] += mij*att*emask/NORM.  Persistent blocks, weights in LDS.
__global__ void __launch_bounds__(128)
gcl_edge_kernel(const float* __restrict__ hsrc, const int* __restrict__ eidx,
                const float* __restrict__ radial, const float* __restrict__ eattr,
                const float* __restrict__ emask,
                const unsigned short* __restrict__ W1, const float* __restrict__ W1x,
                const float* __restrict__ b1,
                const unsigned short* __restrict__ W2, const float* __restrict__ b2,
                const float* __restrict__ Watt, const float* __restrict__ batt,
                float* __restrict__ agg, int nEdges, int nTiles) {
  extern __shared__ unsigned char smem[];
  unsigned short* sW1 = (unsigned short*)smem;       // 256*128 bf16
  unsigned short* sW2 = sW1 + 256 * HID;             // 128*128 bf16
  unsigned short* sA  = sW2 + HID * HID;             // 64*256 bf16 (later sM f32 64*128)
  unsigned short* sH1 = sA + TILE * 256;             // 64*128 bf16
  float* sM    = (float*)sA;
  float* sRad  = (float*)(sH1 + TILE * HID);
  float* sEat  = sRad + TILE;
  float* sAtt  = sEat + TILE;                        // stashes emask, then att*emask/NORM
  int*   sRow  = (int*)(sAtt + TILE);
  float* sWatt = (float*)(sRow + TILE);              // 128 floats, loaded once

  const int tid = threadIdx.x;
  for (int i = tid; i < (256 * HID) / 8; i += blockDim.x) ((U4*)sW1)[i] = ((const U4*)W1)[i];
  for (int i = tid; i < (HID * HID) / 8; i += blockDim.x) ((U4*)sW2)[i] = ((const U4*)W2)[i];
  sWatt[tid] = Watt[tid];                            // blockDim.x == HID

  const int lane = tid & 31, wave = tid >> 5;
  const int half = lane >> 4, ncol = lane & 15;
  const int mBase = wave * 16;

  for (int t = blockIdx.x; t < nTiles; t += gridDim.x) {
    const int e0 = t * TILE;
    __syncthreads();                                 // prev tile fully consumed
    if (tid < TILE) {
      int e = e0 + tid; if (e >= nEdges) e = nEdges - 1;
      sRow[tid] = eidx[e];
      sRad[tid] = radial[e];
      sEat[tid] = eattr[e];
      sAtt[tid] = emask[e];
    }
    for (int g = tid; g < TILE * 64; g += blockDim.x) {   // gather [h_row|h_col] -> bf16
      int m = g >> 6, seg = g & 63;
      int e = e0 + m; if (e >= nEdges) e = nEdges - 1;
      int node = (seg < 32) ? eidx[e] : eidx[nEdges + e];
      const float4 f = *(const float4*)(hsrc + (size_t)node * HID + (seg & 31) * 4);
      unsigned short* dst = sA + m * 256 + ((seg < 32) ? 0 : HID) + (seg & 31) * 4;
      U2 p; p.x = f2bf2(f.x, f.y); p.y = f2bf2(f.z, f.w);
      *(U2*)dst = p;                                  // single 8B LDS store
    }
    __syncthreads();

    // e1 (K=256 via WMMA, 2 ea rows + bias folded into C init), silu -> sH1
    {
      Frag afr[8];
      load_afrags<8, 256>(sA, mBase, lane, afr);     // A resident across all n-tiles
#pragma unroll 1
      for (int nt = 0; nt < 8; ++nt) {
        const int n = nt * 16 + ncol;
        const float bn = b1[n], wr = W1x[n], we = W1x[HID + n];
        v8f acc;
#pragma unroll
        for (int r = 0; r < 8; ++r) {
          int m = mBase + r + 8 * half;
          acc[r] = bn + sRad[m] * wr + sEat[m] * we;
        }
        acc = wmma_row<8, HID>(afr, sW1, nt * 16, lane, acc);
#pragma unroll
        for (int r = 0; r < 8; ++r) {
          int m = mBase + r + 8 * half;
          sH1[m * HID + n] = f2bf(siluf(acc[r]));
        }
      }
    }
    __syncthreads();

    // e2 (K=128), silu -> sM fp32 (overwrites sA)
    {
      Frag afr[4];
      load_afrags<4, HID>(sH1, mBase, lane, afr);
#pragma unroll 1
      for (int nt = 0; nt < 8; ++nt) {
        const int n = nt * 16 + ncol;
        const float bn = b2[n];
        v8f acc;
#pragma unroll
        for (int r = 0; r < 8; ++r) acc[r] = bn;
        acc = wmma_row<4, HID>(afr, sW2, nt * 16, lane, acc);
#pragma unroll
        for (int r = 0; r < 8; ++r) {
          int m = mBase + r + 8 * half;
          sM[m * HID + n] = siluf(acc[r]);
        }
      }
    }
    __syncthreads();

    if (tid < TILE) {                                // attention (rotated -> no bank conflicts)
      float s = batt[0];
      for (int k = 0; k < HID; ++k) {
        int n = (k + tid) & (HID - 1);
        s += sM[tid * HID + n] * sWatt[n];
      }
      sAtt[tid] = sigmf(s) * sAtt[tid] * (1.0f / NORMF);   // fold segment-sum /NORM
    }
    __syncthreads();

    {                                                // scatter: coalesced f32 atomics to L2
      const int n = tid;                             // blockDim.x == HID
      for (int m = 0; m < TILE; ++m)
        if (e0 + m < nEdges)
          atomicAdd(&agg[(size_t)sRow[m] * HID + n], sM[m * HID + n] * sAtt[m]);
    }
  }
}

// ---------------------------------------------------------------------------
// GCL node kernel: hout = (hin + n2(silu(n1([h|agg])))) * node_mask
__global__ void __launch_bounds__(128)
gcl_node_kernel(const float* __restrict__ hin, const float* __restrict__ agg,
                const unsigned short* __restrict__ W1, const float* __restrict__ b1,
                const unsigned short* __restrict__ W2, const float* __restrict__ b2,
                const float* __restrict__ nmask, float* __restrict__ hout,
                int nNodes, int nTiles) {
  extern __shared__ unsigned char smem[];
  unsigned short* sW1 = (unsigned short*)smem;
  unsigned short* sW2 = sW1 + 256 * HID;
  unsigned short* sA  = sW2 + HID * HID;             // 64*256
  unsigned short* sH1 = sA + TILE * 256;             // 64*128

  const int tid = threadIdx.x;
  for (int i = tid; i < (256 * HID) / 8; i += blockDim.x) ((U4*)sW1)[i] = ((const U4*)W1)[i];
  for (int i = tid; i < (HID * HID) / 8; i += blockDim.x) ((U4*)sW2)[i] = ((const U4*)W2)[i];

  const int lane = tid & 31, wave = tid >> 5;
  const int half = lane >> 4, ncol = lane & 15;
  const int mBase = wave * 16;

  for (int t = blockIdx.x; t < nTiles; t += gridDim.x) {
    const int e0 = t * TILE;
    __syncthreads();
    for (int g = tid; g < TILE * 64; g += blockDim.x) {
      int m = g >> 6, seg = g & 63;
      int node = e0 + m; if (node >= nNodes) node = 0;
      const float* src = (seg < 32) ? (hin + (size_t)node * HID + (seg & 31) * 4)
                                    : (agg + (size_t)node * HID + (seg & 31) * 4);
      const float4 f = *(const float4*)src;
      unsigned short* dst = sA + m * 256 + ((seg < 32) ? 0 : HID) + (seg & 31) * 4;
      U2 p; p.x = f2bf2(f.x, f.y); p.y = f2bf2(f.z, f.w);
      *(U2*)dst = p;
    }
    __syncthreads();

    {                                                // n1 + silu
      Frag afr[8];
      load_afrags<8, 256>(sA, mBase, lane, afr);
#pragma unroll 1
      for (int nt = 0; nt < 8; ++nt) {
        const int n = nt * 16 + ncol;
        const float bn = b1[n];
        v8f acc;
#pragma unroll
        for (int r = 0; r < 8; ++r) acc[r] = bn;
        acc = wmma_row<8, HID>(afr, sW1, nt * 16, lane, acc);
#pragma unroll
        for (int r = 0; r < 8; ++r) {
          int m = mBase + r + 8 * half;
          sH1[m * HID + n] = f2bf(siluf(acc[r]));
        }
      }
    }
    __syncthreads();

    {                                                // n2 + residual + mask -> global
      Frag afr[4];
      load_afrags<4, HID>(sH1, mBase, lane, afr);
#pragma unroll 1
      for (int nt = 0; nt < 8; ++nt) {
        const int n = nt * 16 + ncol;
        const float bn = b2[n];
        v8f acc;
#pragma unroll
        for (int r = 0; r < 8; ++r) acc[r] = bn;
        acc = wmma_row<4, HID>(afr, sW2, nt * 16, lane, acc);
#pragma unroll
        for (int r = 0; r < 8; ++r) {
          int node = e0 + mBase + r + 8 * half;
          if (node < nNodes) {
            size_t idx = (size_t)node * HID + n;
            hout[idx] = (hin[idx] + acc[r]) * nmask[node];
          }
        }
      }
    }
  }
}

// ---------------------------------------------------------------------------
// Equivariant edge kernel: s = c3(silu(c2(silu(c1(...))))),
// xagg[row] += cdiff * s * emask / NORM
__global__ void __launch_bounds__(128)
eq_edge_kernel(const float* __restrict__ hsrc, const int* __restrict__ eidx,
               const float* __restrict__ radial, const float* __restrict__ eattr,
               const float* __restrict__ emask, const float* __restrict__ cdiff,
               const unsigned short* __restrict__ W1, const float* __restrict__ W1x,
               const float* __restrict__ b1,
               const unsigned short* __restrict__ W2, const float* __restrict__ b2,
               const float* __restrict__ Wc3,
               float* __restrict__ xagg, int nEdges, int nTiles) {
  extern __shared__ unsigned char smem[];
  unsigned short* sW1 = (unsigned short*)smem;
  unsigned short* sW2 = sW1 + 256 * HID;
  unsigned short* sA  = sW2 + HID * HID;
  unsigned short* sH1 = sA + TILE * 256;
  float* sM   = (float*)sA;
  float* sRad = (float*)(sH1 + TILE * HID);
  float* sEat = sRad + TILE;
  float* sEms = sEat + TILE;
  float* sCd  = sEms + TILE;                         // 3*TILE
  int*   sRow = (int*)(sCd + 3 * TILE);
  float* sWc3 = (float*)(sRow + TILE);               // 128 floats, loaded once

  const int tid = threadIdx.x;
  for (int i = tid; i < (256 * HID) / 8; i += blockDim.x) ((U4*)sW1)[i] = ((const U4*)W1)[i];
  for (int i = tid; i < (HID * HID) / 8; i += blockDim.x) ((U4*)sW2)[i] = ((const U4*)W2)[i];
  sWc3[tid] = Wc3[tid];                              // blockDim.x == HID

  const int lane = tid & 31, wave = tid >> 5;
  const int half = lane >> 4, ncol = lane & 15;
  const int mBase = wave * 16;

  for (int t = blockIdx.x; t < nTiles; t += gridDim.x) {
    const int e0 = t * TILE;
    __syncthreads();
    if (tid < TILE) {
      int e = e0 + tid; if (e >= nEdges) e = nEdges - 1;
      sRow[tid] = eidx[e];
      sRad[tid] = radial[e];
      sEat[tid] = eattr[e];
      sEms[tid] = emask[e];
      sCd[tid * 3 + 0] = cdiff[(size_t)e * 3 + 0];
      sCd[tid * 3 + 1] = cdiff[(size_t)e * 3 + 1];
      sCd[tid * 3 + 2] = cdiff[(size_t)e * 3 + 2];
    }
    for (int g = tid; g < TILE * 64; g += blockDim.x) {
      int m = g >> 6, seg = g & 63;
      int e = e0 + m; if (e >= nEdges) e = nEdges - 1;
      int node = (seg < 32) ? eidx[e] : eidx[nEdges + e];
      const float4 f = *(const float4*)(hsrc + (size_t)node * HID + (seg & 31) * 4);
      unsigned short* dst = sA + m * 256 + ((seg < 32) ? 0 : HID) + (seg & 31) * 4;
      U2 p; p.x = f2bf2(f.x, f.y); p.y = f2bf2(f.z, f.w);
      *(U2*)dst = p;
    }
    __syncthreads();

    {                                                // c1 + silu
      Frag afr[8];
      load_afrags<8, 256>(sA, mBase, lane, afr);
#pragma unroll 1
      for (int nt = 0; nt < 8; ++nt) {
        const int n = nt * 16 + ncol;
        const float bn = b1[n], wr = W1x[n], we = W1x[HID + n];
        v8f acc;
#pragma unroll
        for (int r = 0; r < 8; ++r) {
          int m = mBase + r + 8 * half;
          acc[r] = bn + sRad[m] * wr + sEat[m] * we;
        }
        acc = wmma_row<8, HID>(afr, sW1, nt * 16, lane, acc);
#pragma unroll
        for (int r = 0; r < 8; ++r) {
          int m = mBase + r + 8 * half;
          sH1[m * HID + n] = f2bf(siluf(acc[r]));
        }
      }
    }
    __syncthreads();

    {                                                // c2 + silu -> sM
      Frag afr[4];
      load_afrags<4, HID>(sH1, mBase, lane, afr);
#pragma unroll 1
      for (int nt = 0; nt < 8; ++nt) {
        const int n = nt * 16 + ncol;
        const float bn = b2[n];
        v8f acc;
#pragma unroll
        for (int r = 0; r < 8; ++r) acc[r] = bn;
        acc = wmma_row<4, HID>(afr, sW2, nt * 16, lane, acc);
#pragma unroll
        for (int r = 0; r < 8; ++r) {
          int m = mBase + r + 8 * half;
          sM[m * HID + n] = siluf(acc[r]);
        }
      }
    }
    __syncthreads();

    if (tid < TILE && (e0 + tid) < nEdges) {         // c3 dot + coordinate scatter
      float s = 0.0f;
      for (int k = 0; k < HID; ++k) {
        int n = (k + tid) & (HID - 1);
        s += sM[tid * HID + n] * sWc3[n];
      }
      float tm = s * sEms[tid] * (1.0f / NORMF);
      size_t r = (size_t)sRow[tid] * 3;
      atomicAdd(&xagg[r + 0], sCd[tid * 3 + 0] * tm);
      atomicAdd(&xagg[r + 1], sCd[tid * 3 + 1] * tm);
      atomicAdd(&xagg[r + 2], sCd[tid * 3 + 2] * tm);
    }
  }
}

// ---------------------------------------------------------------------------
__global__ void finalize_h_kernel(float* __restrict__ out_h,
                                  const float* __restrict__ nmask, int nNodes) {
  int i = blockIdx.x * blockDim.x + threadIdx.x;
  if (i < nNodes * HID) out_h[i] *= nmask[i >> 7];
}
__global__ void finalize_x_kernel(const float* __restrict__ x,
                                  const float* __restrict__ xagg,
                                  const float* __restrict__ nmask,
                                  float* __restrict__ out_x, int nNodes) {
  int i = blockIdx.x * blockDim.x + threadIdx.x;
  if (i >= nNodes) return;
  float mk = nmask[i];
  out_x[i * 3 + 0] = (x[i * 3 + 0] + xagg[i * 3 + 0]) * mk;
  out_x[i * 3 + 1] = (x[i * 3 + 1] + xagg[i * 3 + 1]) * mk;
  out_x[i * 3 + 2] = (x[i * 3 + 2] + xagg[i * 3 + 2]) * mk;
}

// ---------------------------------------------------------------------------
extern "C" void kernel_launch(void* const* d_in, const int* in_sizes, int n_in,
                              void* d_out, int out_size, void* d_ws, size_t ws_size,
                              hipStream_t stream) {
  const float* h     = (const float*)d_in[0];
  const float* x     = (const float*)d_in[1];
  const int*   eidx  = (const int*)d_in[2];
  const float* nmask = (const float*)d_in[3];
  const float* emask = (const float*)d_in[4];
  const float* eattr = (const float*)d_in[5];
  const int N = in_sizes[3];          // node_mask count
  const int E = in_sizes[4];          // edge_mask count

  // params in setup_inputs() insertion order:
  // gcl0: e1.W e1.b e2.W e2.b n1.W n1.b n2.W n2.b att.W att.b   -> P[0..9]
  // gcl1: same                                                  -> P[10..19]
  // eq:   c1.W c1.b c2.W c2.b c3.W                              -> P[20..24]
  const float* P[25];
  for (int i = 0; i < 25; ++i) P[i] = (const float*)d_in[6 + i];

  // ---- workspace carve ----
  char* w = (char*)d_ws;
  auto alloc = [&](size_t bytes) -> char* {
    char* p = w; w += (bytes + 255) & ~(size_t)255; return p;
  };
  float* radial = (float*)alloc((size_t)E * 4);
  float* cdiff  = (float*)alloc((size_t)E * 12);
  float* h1     = (float*)alloc((size_t)N * HID * 4);
  float* agg    = (float*)alloc((size_t)N * HID * 4);
  float* xagg   = (float*)alloc((size_t)N * 12);
  unsigned short *bwE1[2], *bwE2[2], *bwN1[2], *bwN2[2];
  for (int l = 0; l < 2; ++l) {
    bwE1[l] = (unsigned short*)alloc(256 * HID * 2);
    bwE2[l] = (unsigned short*)alloc(HID * HID * 2);
    bwN1[l] = (unsigned short*)alloc(256 * HID * 2);
    bwN2[l] = (unsigned short*)alloc(HID * HID * 2);
  }
  unsigned short* bwC1 = (unsigned short*)alloc(256 * HID * 2);
  unsigned short* bwC2 = (unsigned short*)alloc(HID * HID * 2);

  // ---- dynamic LDS sizes ----
  const int GEMM_USH   = (256 * HID + HID * HID + TILE * 256 + TILE * HID) * 2; // 147456 B
  const int GCL_SMEM   = GEMM_USH + (4 * TILE) * 4 + HID * 4;                   // +meta +Watt
  const int EQ_SMEM    = GEMM_USH + (7 * TILE) * 4 + HID * 4;                   // +meta +Wc3
  const int NODE_SMEM  = GEMM_USH;
  (void)hipFuncSetAttribute(reinterpret_cast<const void*>(gcl_edge_kernel),
                            hipFuncAttributeMaxDynamicSharedMemorySize, GCL_SMEM);
  (void)hipFuncSetAttribute(reinterpret_cast<const void*>(eq_edge_kernel),
                            hipFuncAttributeMaxDynamicSharedMemorySize, EQ_SMEM);
  (void)hipFuncSetAttribute(reinterpret_cast<const void*>(gcl_node_kernel),
                            hipFuncAttributeMaxDynamicSharedMemorySize, NODE_SMEM);

  // ---- weight conversion (fp32 -> bf16) ----
  auto conv = [&](const float* s, unsigned short* d, int n) {
    f2bf_kernel<<<(n + 255) / 256, 256, 0, stream>>>(s, d, n);
  };
  for (int l = 0; l < 2; ++l) {
    int b = l * 10;
    conv(P[b + 0], bwE1[l], 256 * HID);   // e1.W first 256 rows (h_row|h_col part)
    conv(P[b + 2], bwE2[l], HID * HID);
    conv(P[b + 4], bwN1[l], 256 * HID);
    conv(P[b + 6], bwN2[l], HID * HID);
  }
  conv(P[20], bwC1, 256 * HID);
  conv(P[22], bwC2, HID * HID);

  // ---- pipeline ----
  prep_edges_kernel<<<(E + 255) / 256, 256, 0, stream>>>(x, eidx, radial, cdiff, E);
  hipMemsetAsync(agg, 0, (size_t)N * HID * 4, stream);
  hipMemsetAsync(xagg, 0, (size_t)N * 12, stream);

  const int eTiles = (E + TILE - 1) / TILE;
  const int nTiles = (N + TILE - 1) / TILE;
  const int eBlocks = eTiles < 1024 ? eTiles : 1024;
  const int nBlocks = nTiles < 1024 ? nTiles : 1024;

  float* out_h = (float*)d_out;
  float* out_x = out_h + (size_t)N * HID;

  // GCL 0
  gcl_edge_kernel<<<eBlocks, 128, GCL_SMEM, stream>>>(
      h, eidx, radial, eattr, emask,
      bwE1[0], P[0] + 256 * HID, P[1], bwE2[0], P[3], P[8], P[9],
      agg, E, eTiles);
  gcl_node_kernel<<<nBlocks, 128, NODE_SMEM, stream>>>(
      h, agg, bwN1[0], P[5], bwN2[0], P[7], nmask, h1, N, nTiles);

  // GCL 1
  hipMemsetAsync(agg, 0, (size_t)N * HID * 4, stream);
  gcl_edge_kernel<<<eBlocks, 128, GCL_SMEM, stream>>>(
      h1, eidx, radial, eattr, emask,
      bwE1[1], P[10] + 256 * HID, P[11], bwE2[1], P[13], P[18], P[19],
      agg, E, eTiles);
  gcl_node_kernel<<<nBlocks, 128, NODE_SMEM, stream>>>(
      h1, agg, bwN1[1], P[15], bwN2[1], P[17], nmask, out_h, N, nTiles);

  // Equivariant coordinate update
  eq_edge_kernel<<<eBlocks, 128, EQ_SMEM, stream>>>(
      out_h, eidx, radial, eattr, emask, cdiff,
      bwC1, P[20] + 256 * HID, P[21], bwC2, P[23], P[24],
      xagg, E, eTiles);

  finalize_h_kernel<<<(N * HID + 255) / 256, 256, 0, stream>>>(out_h, nmask, N);
  finalize_x_kernel<<<(N + 255) / 256, 256, 0, stream>>>(x, xagg, nmask, out_x, N);
}
